// GATConv_39840116638416
// MI455X (gfx1250) — matrix operand order repaired and checked by reading, not compiled, not run on hardware
//
#include <hip/hip_runtime.h>
#include <hip/hip_bf16.h>

// GAT layer, fused flash-style for MI455X (gfx1250, wave32, WMMA).
// B=128, N=512, C_IN=C_OUT=64.
// ws layout (bytes): h0 f32 [0, 16MiB) | h1^T f16 [16MiB, 24MiB) | s1 | s2

#define GAT_B     128
#define GAT_N     512
#define GAT_C     64
#define GAT_ALPHA 0.2f
#define GAT_NEGINF (-9000000000000000.0f)

typedef __attribute__((ext_vector_type(16))) _Float16 v16h;
typedef __attribute__((ext_vector_type(8)))  _Float16 v8h;
typedef __attribute__((ext_vector_type(8)))  float    v8f;
typedef __attribute__((ext_vector_type(4)))  float    v4f;
typedef __attribute__((ext_vector_type(4)))  int      v4i;

__device__ __forceinline__ v8f wmma_f16(v16h a, v16h b, v8f c) {
  // D = A(16x32 f16) * B(32x16 f16) + C(16x16 f32)
  return __builtin_amdgcn_wmma_f32_16x16x32_f16(
      /*neg_a=*/false, a, /*neg_b=*/false, b,
      /*c_mod=*/(short)0, c, /*reuse_a=*/false, /*reuse_b=*/false);
}

// Async memory->LDS b128 transfer (per active lane), tracked by ASYNCcnt.
__device__ __forceinline__ void async_g2l_b128(const void* gptr, void* lptr) {
#if defined(__has_builtin) && __has_builtin(__builtin_amdgcn_global_load_async_to_lds_b128)
  typedef __attribute__((address_space(1))) v4i g_v4i;   // global int4
  typedef __attribute__((address_space(3))) v4i l_v4i;   // LDS int4
  __builtin_amdgcn_global_load_async_to_lds_b128(
      (g_v4i*)gptr, (l_v4i*)lptr, 0, 0);
#else
  unsigned int l32 = (unsigned int)(unsigned long)(lptr);
  asm volatile("global_load_async_to_lds_b128 %0, %1, off"
               :: "v"(l32), "v"((unsigned long)gptr) : "memory");
#endif
}

__device__ __forceinline__ void wait_asynccnt0() {
#if defined(__has_builtin) && __has_builtin(__builtin_amdgcn_s_wait_asynccnt)
  __builtin_amdgcn_s_wait_asynccnt(0);
#else
  asm volatile("s_wait_asynccnt 0x0" ::: "memory");
#endif
}

// ---------------------------------------------------------------------------
// Kernel 1: h0 = x@W0, h1 = x@W1 (WMMA f16), h = h0+h1, s1 = h@a1, s2 = h@a2.
// One wave owns a 16-row x-tile; 256 threads = 8 waves = 128 rows/block.
// ---------------------------------------------------------------------------
__global__ __launch_bounds__(256) void gat_proj(
    const float* __restrict__ x,      // [B*N, 64]
    const float* __restrict__ W,      // [2, 64, 64]
    const float* __restrict__ av,     // [128]
    float*    __restrict__ h0,        // [B*N, 64]
    _Float16* __restrict__ h1t,       // [B, 64, 512]  (feature-major)
    float*    __restrict__ s1,        // [B*N]
    float*    __restrict__ s2)        // [B*N]
{
  __shared__ _Float16 wt[2][64][64];  // W transposed: wt[w][f][c], f16 (16 KB)

  const int tid = threadIdx.x;
  // Stage both weight matrices transposed + converted to f16.
  #pragma unroll
  for (int w = 0; w < 2; ++w) {
    #pragma unroll
    for (int k = 0; k < 16; ++k) {
      int idx = tid * 16 + k;            // 0..4095 over (c,f)
      int c = idx >> 6, f = idx & 63;
      wt[w][f][c] = (_Float16)W[w * 4096 + idx];
    }
  }
  __syncthreads();

  const int wave = tid >> 5;
  const int lane = tid & 31;
  const int g    = lane >> 4;            // half-wave group
  const int m    = lane & 15;
  const long Rbase = (long)blockIdx.x * 128 + wave * 16;
  const long row   = Rbase + m;          // A-matrix row for this lane

  // Build A tiles (16x32 f16): lane half-group g holds
  // K in {16g..16g+15}: one contiguous 16-float run of x.
  v16h A[2];
  #pragma unroll
  for (int kb = 0; kb < 2; ++kb) {
    const float* xp = x + row * 64 + kb * 32 + g * 16;
    v4f x0 = *(const v4f*)(xp);
    v4f x1 = *(const v4f*)(xp + 4);
    v4f x2 = *(const v4f*)(xp + 8);
    v4f x3 = *(const v4f*)(xp + 12);
    #pragma unroll
    for (int i = 0; i < 4; ++i) {
      A[kb][i]      = (_Float16)x0[i];
      A[kb][4 + i]  = (_Float16)x1[i];
      A[kb][8 + i]  = (_Float16)x2[i];
      A[kb][12 + i] = (_Float16)x3[i];
    }
  }

  v8f acc0[4], acc1[4];
  #pragma unroll
  for (int ct = 0; ct < 4; ++ct) {
    #pragma unroll
    for (int r = 0; r < 8; ++r) { acc0[ct][r] = 0.f; acc1[ct][r] = 0.f; }
  }

  // 2 K-blocks x 4 col-tiles x 2 weight matrices = 16 WMMAs per wave.
  #pragma unroll
  for (int kb = 0; kb < 2; ++kb) {
    const int kbase = kb * 32 + g * 16;  // B layout: lanes 16-31 hold K+16
    #pragma unroll
    for (int ct = 0; ct < 4; ++ct) {
      const int col = ct * 16 + m;
      v16h B0 = *(const v16h*)&wt[0][col][kbase];  // 32B contiguous LDS
      v16h B1 = *(const v16h*)&wt[1][col][kbase];
      acc0[ct] = wmma_f16(A[kb], B0, acc0[ct]);
      acc1[ct] = wmma_f16(A[kb], B1, acc1[ct]);
    }
  }

  // s1/s2: dot(h, a1/a2) with cross-lane butterfly inside each 16-lane half.
  float a1v[4], a2v[4];
  #pragma unroll
  for (int ct = 0; ct < 4; ++ct) {
    a1v[ct] = av[ct * 16 + m];
    a2v[ct] = av[64 + ct * 16 + m];
  }
  float p1[8], p2[8];
  #pragma unroll
  for (int r = 0; r < 8; ++r) {
    float h1s = 0.f, h2s = 0.f;
    #pragma unroll
    for (int ct = 0; ct < 4; ++ct) {
      float hv = acc0[ct][r] + acc1[ct][r];
      h1s += hv * a1v[ct];
      h2s += hv * a2v[ct];
    }
    #pragma unroll
    for (int sh = 1; sh < 16; sh <<= 1) {
      h1s += __shfl_xor(h1s, sh, 32);
      h2s += __shfl_xor(h2s, sh, 32);
    }
    p1[r] = h1s; p2[r] = h2s;
  }
  if (m == 0) {                       // lane 0 -> rows 0..7, lane 16 -> rows 8..15
    #pragma unroll
    for (int r = 0; r < 8; ++r) {
      long rr = Rbase + g * 8 + r;
      s1[rr] = p1[r];
      s2[rr] = p2[r];
    }
  }

  // Store h0 (f32, row-major, per-element) and h1 (f16, transposed [b][f][n]).
  // The 8 C rows of this lane are consecutive nodes -> one b128 store per ct.
  const long bb = Rbase >> 9;                    // batch (block never straddles)
  const int  n0 = (int)(Rbase & 511) + g * 8;    // first node of this lane's rows
  #pragma unroll
  for (int ct = 0; ct < 4; ++ct) {
    const int f = ct * 16 + m;
    v8h hp;
    #pragma unroll
    for (int r = 0; r < 8; ++r) {
      long rr = Rbase + g * 8 + r;               // C layout: M = r + 8*(lane/16)
      h0[rr * 64 + f] = acc0[ct][r];
      hp[r] = (_Float16)acc1[ct][r];
    }
    *(v8h*)(h1t + (bb * 64 + f) * 512 + n0) = hp;
  }
}

// ---------------------------------------------------------------------------
// Kernel 2: fused masked softmax + P @ h1 (WMMA) + diag*h0 + bias.
// Block = 128 threads (4 waves) = 64 rows of one batch. LDS: h1^T[b] 64KB,
// P 64KB, s2 2KB, diag 256B  ->  131 KB (2 workgroups / 320KB WGP).
// ---------------------------------------------------------------------------
__global__ __launch_bounds__(128) void gat_attn(
    const int*      __restrict__ adj,   // [512, 512]
    const float*    __restrict__ bias,  // [64]
    const float*    __restrict__ h0,    // [B*N, 64]
    const _Float16* __restrict__ h1t,   // [B, 64, 512]
    const float*    __restrict__ s1,    // [B*N]
    const float*    __restrict__ s2,    // [B*N]
    float*          __restrict__ out)   // [B*N, 64]
{
  extern __shared__ char smem[];
  _Float16* h1s   = (_Float16*)(smem);            // [64][512] f16
  _Float16* Ps    = (_Float16*)(smem + 65536);    // [64][512] f16 probabilities
  float*    s2s   = (float*)(smem + 131072);      // [512]
  float*    dstat = (float*)(smem + 133120);      // [64] softmax diag per row

  const int tid  = threadIdx.x;
  const int wave = tid >> 5, lane = tid & 31;
  const int g    = lane >> 4, m = lane & 15;
  const int b    = blockIdx.x >> 3;
  const int nb   = (blockIdx.x & 7) * 64;         // node base of this block

  // Stage h1^T[b] (64x512 f16 = 64KB) into LDS with async b128 transfers
  // (memory -> LDS directly, no VGPR round-trip; drained via ASYNCcnt).
  {
    const uint4* src = (const uint4*)(h1t + (long)b * 64 * 512);
    uint4* dst = (uint4*)h1s;
    #pragma unroll 4
    for (int i = tid; i < 4096; i += 128)
      async_g2l_b128(src + i, dst + i);
    const float* s2g = s2 + (long)b * 512;
    for (int i = tid; i < 512; i += 128) s2s[i] = s2g[i];
    wait_asynccnt0();
  }
  __syncthreads();

  const int rowLocal0 = wave * 16;

  // ---- Pass A: masked leaky-relu scores -> softmax -> P in LDS ----
  for (int i = 0; i < 16; ++i) {
    const int ni  = nb + rowLocal0 + i;           // node index in [0,512)
    const float s1i = s1[(long)b * 512 + ni];
    if (i < 15)  // prefetch next adj row (global_prefetch_b8)
      __builtin_prefetch(adj + (ni + 1) * 512 + lane, 0, 1);
    float ev[16];
    float mx = -3.4e38f;
    #pragma unroll
    for (int t = 0; t < 16; ++t) {
      const int j = lane + 32 * t;
      const int av = adj[ni * 512 + j];
      float e = s1i + s2s[j];
      e = (e > 0.f) ? e : GAT_ALPHA * e;          // leaky_relu
      e = (av > 0) ? e : GAT_NEGINF;              // adjacency mask
      ev[t] = e;
      mx = fmaxf(mx, e);
    }
    #pragma unroll
    for (int sh = 1; sh < 32; sh <<= 1) mx = fmaxf(mx, __shfl_xor(mx, sh, 32));
    float sum = 0.f;
    #pragma unroll
    for (int t = 0; t < 16; ++t) {
      float p = __expf(ev[t] - mx);
      ev[t] = p;
      sum += p;
    }
    #pragma unroll
    for (int sh = 1; sh < 32; sh <<= 1) sum += __shfl_xor(sum, sh, 32);
    const float inv = 1.f / sum;

    _Float16* prow = Ps + (rowLocal0 + i) * 512;
    #pragma unroll
    for (int t = 0; t < 16; ++t) {
      const int j = lane + 32 * t;
      float p = ev[t] * inv;
      if (j == ni) {                               // diag: save, then zero (1-eye)
        dstat[rowLocal0 + i] = p;
        p = 0.f;
      }
      prow[j] = (_Float16)p;
    }
  }
  // P rows and dstat are produced and consumed by the same wave; LDS ops from
  // one wave are in-order (DScnt), so no extra barrier needed here.

  // ---- Pass B: out_tile = P(16x512) @ h1(512x64), 64 WMMAs per wave ----
  v8f acc[4];
  #pragma unroll
  for (int ct = 0; ct < 4; ++ct) {
    #pragma unroll
    for (int r = 0; r < 8; ++r) acc[ct][r] = 0.f;
  }

  const _Float16* pArow = Ps + (rowLocal0 + m) * 512;
  for (int kb = 0; kb < 16; ++kb) {
    // A: 16x32 f16 layout -> one contiguous 16-half run per lane (K = 16g + s).
    const int j0 = kb * 32 + g * 16;
    v8h a0 = *(const v8h*)(pArow + j0);
    v8h a1 = *(const v8h*)(pArow + j0 + 8);
    v16h Am;
    #pragma unroll
    for (int i2 = 0; i2 < 8; ++i2) { Am[i2] = a0[i2]; Am[i2 + 8] = a1[i2]; }

    const int kcol = kb * 32 + g * 16;            // B: lanes 16-31 hold K+16
    #pragma unroll
    for (int ct = 0; ct < 4; ++ct) {
      const v16h Bm = *(const v16h*)(h1s + (ct * 16 + m) * 512 + kcol);
      acc[ct] = wmma_f16(Am, Bm, acc[ct]);
    }
  }

  // ---- Epilogue: + diag * h0 + bias, store ----
  #pragma unroll
  for (int ct = 0; ct < 4; ++ct) {
    const int f = ct * 16 + m;
    const float bv = bias[f];
    #pragma unroll
    for (int r = 0; r < 8; ++r) {
      const int mrow = r + 8 * g;                 // C layout row
      const int ni = nb + rowLocal0 + mrow;
      const long off = ((long)b * 512 + ni) * 64 + f;
      out[off] = acc[ct][r] + dstat[rowLocal0 + mrow] * h0[off] + bv;
    }
  }
}

// ---------------------------------------------------------------------------
extern "C" void kernel_launch(void* const* d_in, const int* in_sizes, int n_in,
                              void* d_out, int out_size, void* d_ws, size_t ws_size,
                              hipStream_t stream) {
  const float* x    = (const float*)d_in[0];   // (128,512,64) f32
  const int*   adj  = (const int*)  d_in[1];   // (512,512) i32
  const float* W    = (const float*)d_in[2];   // (2,64,64) f32
  const float* av   = (const float*)d_in[3];   // (128,1) f32
  const float* bias = (const float*)d_in[4];   // (64,) f32
  float* out = (float*)d_out;                  // (128,512,64) f32

  char* ws = (char*)d_ws;                      // ~24.5 MiB used
  float*    h0  = (float*)(ws);                          // 16,777,216 B
  _Float16* h1t = (_Float16*)(ws + 16777216);            //  8,388,608 B
  float*    s1  = (float*)(ws + 25165824);               //    262,144 B
  float*    s2  = (float*)(ws + 25427968);               //    262,144 B

  // Opt in to >64KB dynamic LDS for the fused attention kernel (320KB/WGP).
  (void)hipFuncSetAttribute((const void*)gat_attn,
                            hipFuncAttributeMaxDynamicSharedMemorySize, 133376);

  // 65536 rows / 128 rows-per-block = 512 blocks.
  gat_proj<<<512, 256, 0, stream>>>(x, W, av, h0, h1t, s1, s2);
  // 128 batches x 8 row-blocks of 64 = 1024 blocks; 131 KB dynamic LDS.
  gat_attn<<<1024, 128, 133376, stream>>>(adj, bias, h0, h1t, s1, s2, out);
}
